// RPN_21921513079207
// MI455X (gfx1250) — matrix operand order repaired and checked by reading, not compiled, not run on hardware
//
#include <hip/hip_runtime.h>
#include <hip/hip_fp16.h>
#include <stdint.h>

typedef __attribute__((ext_vector_type(16))) _Float16 v16h;
typedef __attribute__((ext_vector_type(8)))  float    v8f;

#define Hdim 100
#define Wdim 100
#define NPIX 10000
#define C 512
#define NA 9
#define NBOX 90000
#define NLOC 36
#define NSCR 18
#define PRE_NMS 12000
#define POST_NMS 2000
#define NMS_THRESH 0.7f
#define MASK_WORDS 375          // 12000 / 32

// d_out offsets (in floats): rpn_locs, rpn_scores, roi, anchors (concatenated flat)
#define OUT_LOCS   0
#define OUT_SCORES 360000
#define OUT_ROI    540000
#define OUT_ANCH   548000

// ---------------- convert / pack kernels ----------------

// x: NCHW f32 [512][100][100] -> xh: NHWC f16 [10000][512]
__global__ void k_convert_x(const float* __restrict__ x, _Float16* __restrict__ xh) {
  int t = blockIdx.x * blockDim.x + threadIdx.x;
  if (t >= NPIX * C) return;
  int pix = t >> 9, cin = t & 511;
  xh[t] = (_Float16)x[cin * NPIX + pix];
}

// w1 OIHW f32 [512][512][3][3] -> fragment-major f16:
// whp[ ((tap*16 + kc)*32 + nc)*512 + lane*16 + e ]
//   = w1[n = nc*16 + (lane&15)][k = kc*32 + 8*(lane>>4) + (e&7) + 16*(e>>3)][ky][kx]
__global__ void k_pack_w1(const float* __restrict__ w1, _Float16* __restrict__ whp) {
  int t = blockIdx.x * blockDim.x + threadIdx.x;
  if (t >= 9 * 16 * 32 * 512) return;
  int e    = t & 15;
  int lane = (t >> 4) & 31;
  int r    = t >> 9;
  int nc   = r & 31; r >>= 5;
  int kc   = r & 15;
  int tap  = r >> 4;
  int k = kc * 32 + ((lane >> 4) * 8) + (e & 7) + ((e >> 3) * 16);
  int n = nc * 16 + (lane & 15);
  int ky = tap / 3, kx = tap % 3;
  whp[t] = (_Float16)w1[((n * C + k) * 3 + ky) * 3 + kx];
}

// Head weights: Whead[k][n] (K=512, N=64: 0..35 w_loc, 36..53 w_score, rest 0)
__global__ void k_pack_whead(const float* __restrict__ wl, const float* __restrict__ wsc,
                             _Float16* __restrict__ whd) {
  int t = blockIdx.x * blockDim.x + threadIdx.x;
  if (t >= 16 * 4 * 512) return;
  int e    = t & 15;
  int lane = (t >> 4) & 31;
  int r    = t >> 9;
  int nc   = r & 3;
  int kc   = r >> 2;
  int k = kc * 32 + ((lane >> 4) * 8) + (e & 7) + ((e >> 3) * 16);
  int n = nc * 16 + (lane & 15);
  float v = 0.0f;
  if (n < NLOC)             v = wl[n * C + k];
  else if (n < NLOC + NSCR) v = wsc[(n - NLOC) * C + k];
  whd[t] = (_Float16)v;
}

// ---------------- 3x3 conv as WMMA implicit GEMM ----------------
// Block: 256 threads = 8 waves. Tile: M=128 positions x N=64 out channels.
__global__ __launch_bounds__(256) void k_conv(const _Float16* __restrict__ xh,
                                              const _Float16* __restrict__ whp,
                                              const float* __restrict__ bias1,
                                              _Float16* __restrict__ h1h) {
  __shared__ _Float16 Bs[8 * 4 * 512];  // 32 KB
  const int tid   = threadIdx.x;
  const int wave  = tid >> 5;
  const int lane  = tid & 31;
  const int mtile = blockIdx.x;   // 0..78
  const int ntile = blockIdx.y;   // 0..7
  const int m0    = mtile * 128 + wave * 16;
  const int khalf = lane >> 4;
  const int mrow  = lane & 15;
  const int p     = m0 + mrow;
  const int pr    = p / Wdim, pc = p % Wdim;

  v8f acc0 = {}, acc1 = {}, acc2 = {}, acc3 = {};

  for (int tap = 0; tap < 9; ++tap) {
    const int ky = tap / 3, kx = tap % 3;
    const int sr = pr + ky - 1, sc = pc + kx - 1;
    const bool avalid = (p < NPIX) && ((unsigned)sr < (unsigned)Hdim) &&
                        ((unsigned)sc < (unsigned)Wdim);
    const size_t pixbase = (size_t)(sr * Wdim + sc) * C + 8 * khalf;

    for (int kh = 0; kh < 2; ++kh) {
      __syncthreads();
#pragma unroll
      for (int u = 0; u < 8; ++u) {
        int v = tid + u * 256;
        int d = v * 8;
        int chunk = d >> 9;
        int cg = (tap * 16 + kh * 8 + (chunk >> 2)) * 32 + ntile * 4 + (chunk & 3);
        *(uint4*)(&Bs[d]) = *(const uint4*)(&whp[(size_t)cg * 512 + (d & 511)]);
      }
      __syncthreads();

      for (int kc2 = 0; kc2 < 8; ++kc2) {
        const int kb = (kh * 8 + kc2) * 32;
        union { uint4 q[2]; v16h h; } a;
        if (avalid) {
          a.q[0] = *(const uint4*)(&xh[pixbase + kb]);
          a.q[1] = *(const uint4*)(&xh[pixbase + kb + 16]);
        } else {
          a.q[0] = make_uint4(0u, 0u, 0u, 0u);
          a.q[1] = make_uint4(0u, 0u, 0u, 0u);
        }
        const _Float16* bp = &Bs[(kc2 * 4) * 512 + lane * 16];
        v16h b0 = *(const v16h*)(bp);
        v16h b1 = *(const v16h*)(bp + 512);
        v16h b2 = *(const v16h*)(bp + 1024);
        v16h b3 = *(const v16h*)(bp + 1536);
        acc0 = __builtin_amdgcn_wmma_f32_16x16x32_f16(false, a.h, false, b0, (short)0, acc0, false, false);
        acc1 = __builtin_amdgcn_wmma_f32_16x16x32_f16(false, a.h, false, b1, (short)0, acc1, false, false);
        acc2 = __builtin_amdgcn_wmma_f32_16x16x32_f16(false, a.h, false, b2, (short)0, acc2, false, false);
        acc3 = __builtin_amdgcn_wmma_f32_16x16x32_f16(false, a.h, false, b3, (short)0, acc3, false, false);
      }
    }
  }

  const int nbase = ntile * 64;
  const int nlane = lane & 15;
  const int moff  = 8 * khalf;
  v8f accs[4] = {acc0, acc1, acc2, acc3};
#pragma unroll
  for (int nc = 0; nc < 4; ++nc) {
    int n = nbase + nc * 16 + nlane;
    float b = bias1[n];
#pragma unroll
    for (int rr = 0; rr < 8; ++rr) {
      int pp = m0 + rr + moff;
      if (pp < NPIX) {
        float v = accs[nc][rr] + b;
        v = v > 0.0f ? v : 0.0f;
        h1h[(size_t)pp * C + n] = (_Float16)v;
      }
    }
  }
}

// ---------------- 1x1 head GEMM (locs + scores fused, N=64 pad) ----------------
__global__ __launch_bounds__(256) void k_head(const _Float16* __restrict__ h1h,
                                              const _Float16* __restrict__ whd,
                                              const float* __restrict__ bl,
                                              const float* __restrict__ bs,
                                              float* __restrict__ out) {
  const int tid = threadIdx.x, wave = tid >> 5, lane = tid & 31;
  const int mtile = blockIdx.x * 8 + wave;
  if (mtile >= NPIX / 16) return;
  const int m0 = mtile * 16;
  const int khalf = lane >> 4, mrow = lane & 15;
  const size_t abase = (size_t)(m0 + mrow) * C + 8 * khalf;

  v8f acc0 = {}, acc1 = {}, acc2 = {}, acc3 = {};
  for (int kc = 0; kc < 16; ++kc) {
    union { uint4 q[2]; v16h h; } a;
    a.q[0] = *(const uint4*)(&h1h[abase + kc * 32]);
    a.q[1] = *(const uint4*)(&h1h[abase + kc * 32 + 16]);
    const _Float16* bp = &whd[(kc * 4) * 512 + lane * 16];
    v16h b0 = *(const v16h*)(bp);
    v16h b1 = *(const v16h*)(bp + 512);
    v16h b2 = *(const v16h*)(bp + 1024);
    v16h b3 = *(const v16h*)(bp + 1536);
    acc0 = __builtin_amdgcn_wmma_f32_16x16x32_f16(false, a.h, false, b0, (short)0, acc0, false, false);
    acc1 = __builtin_amdgcn_wmma_f32_16x16x32_f16(false, a.h, false, b1, (short)0, acc1, false, false);
    acc2 = __builtin_amdgcn_wmma_f32_16x16x32_f16(false, a.h, false, b2, (short)0, acc2, false, false);
    acc3 = __builtin_amdgcn_wmma_f32_16x16x32_f16(false, a.h, false, b3, (short)0, acc3, false, false);
  }

  const int nlane = lane & 15, moff = 8 * khalf;
  v8f accs[4] = {acc0, acc1, acc2, acc3};
#pragma unroll
  for (int nc = 0; nc < 4; ++nc) {
    int n = nc * 16 + nlane;
#pragma unroll
    for (int rr = 0; rr < 8; ++rr) {
      int pp = m0 + rr + moff;
      float v = accs[nc][rr];
      if (n < NLOC) {
        out[OUT_LOCS + (size_t)pp * NLOC + n] = v + bl[n];
      } else if (n < NLOC + NSCR) {
        int o = n - NLOC;
        out[OUT_SCORES + (size_t)pp * NSCR + o] = v + bs[o];
      }
    }
  }
}

// ---------------- anchors ----------------
__global__ void k_anchors(float* __restrict__ out) {
  int t = blockIdx.x * blockDim.x + threadIdx.x;
  if (t >= NBOX) return;
  int pos = t / NA, a = t % NA;
  int ri = a / 3, si = a % 3;
  const float ratios[3] = {0.5f, 1.0f, 2.0f};
  const float scales[3] = {8.0f, 16.0f, 32.0f};
  float r = ratios[ri], s = scales[si];
  float hh = 16.0f * s * __builtin_sqrtf(r);
  float ww = 16.0f * s * __builtin_sqrtf(1.0f / r);
  float sx = (float)(pos % Wdim) * 16.0f;
  float sy = (float)(pos / Wdim) * 16.0f;
  float* o = out + OUT_ANCH + (size_t)t * 4;
  o[0] = sx - 0.5f * ww;
  o[1] = sy - 0.5f * hh;
  o[2] = sx + 0.5f * ww;
  o[3] = sy + 0.5f * hh;
}

__global__ void k_zero_roi(float* __restrict__ out) {
  int t = blockIdx.x * blockDim.x + threadIdx.x;
  if (t < POST_NMS * 4) out[OUT_ROI + t] = 0.0f;
}

// ---------------- decode + clip + min-size mask ----------------
__global__ void k_decode(const float* __restrict__ out, const int* __restrict__ ih,
                         const int* __restrict__ iw, const int* __restrict__ isc,
                         float* __restrict__ boxes, float* __restrict__ key) {
  int i = blockIdx.x * blockDim.x + threadIdx.x;
  if (i >= NBOX) return;
  const float* an = out + OUT_ANCH + (size_t)i * 4;
  const float* lc = out + OUT_LOCS + (size_t)i * 4;
  float score = out[OUT_SCORES + (size_t)i * 2 + 1];
  float aw = an[2] - an[0], ah = an[3] - an[1];
  float ax = an[0] + 0.5f * aw, ay = an[1] + 0.5f * ah;
  float cx = lc[0] * aw + ax, cy = lc[1] * ah + ay;
  float bw = expf(lc[2]) * aw, bh = expf(lc[3]) * ah;
  float imw = (float)iw[0], imh = (float)ih[0];
  float x1 = fminf(fmaxf(cx - 0.5f * bw, 0.0f), imw);
  float y1 = fminf(fmaxf(cy - 0.5f * bh, 0.0f), imh);
  float x2 = fminf(fmaxf(cx + 0.5f * bw, 0.0f), imw);
  float y2 = fminf(fmaxf(cy + 0.5f * bh, 0.0f), imh);
  float minsz = 16.0f * (float)isc[0];
  bool valid = ((x2 - x1) >= minsz) && ((y2 - y1) >= minsz);
  boxes[(size_t)i * 4 + 0] = x1;
  boxes[(size_t)i * 4 + 1] = y1;
  boxes[(size_t)i * 4 + 2] = x2;
  boxes[(size_t)i * 4 + 3] = y2;
  key[i] = valid ? score : -__builtin_inff();
}

// ---------------- exact stable top-PRE_NMS via LDS-tiled rank counting ----------------
__global__ __launch_bounds__(256) void k_rank(const float* __restrict__ key,
                                              const float* __restrict__ boxes,
                                              float* __restrict__ sboxes,
                                              int* __restrict__ svalid) {
  __shared__ float tk[2048];
  const int i = blockIdx.x * 256 + threadIdx.x;
  const float ki = (i < NBOX) ? key[i] : 0.0f;
  int rank = 0;
  for (int t0 = 0; t0 < NBOX; t0 += 2048) {
    int nt = NBOX - t0; if (nt > 2048) nt = 2048;
    __syncthreads();
    for (int u = threadIdx.x; u < 2048; u += 256) {
      int j = t0 + u;
      tk[u] = (j < NBOX) ? key[j] : 0.0f;
    }
    __syncthreads();
    for (int jj = 0; jj < nt; ++jj) {
      float kj = tk[jj];
      int j = t0 + jj;
      rank += (int)((kj > ki) || (kj == ki && j < i));
    }
  }
  if (i < NBOX && rank < PRE_NMS) {
    sboxes[(size_t)rank * 4 + 0] = boxes[(size_t)i * 4 + 0];
    sboxes[(size_t)rank * 4 + 1] = boxes[(size_t)i * 4 + 1];
    sboxes[(size_t)rank * 4 + 2] = boxes[(size_t)i * 4 + 2];
    sboxes[(size_t)rank * 4 + 3] = boxes[(size_t)i * 4 + 3];
    svalid[rank] = (ki != -__builtin_inff()) ? 1 : 0;
  }
}

// ---------------- phase A: suppression bit-matrix (parallel) ----------------
__global__ void k_mask(const float* __restrict__ sboxes, uint32_t* __restrict__ mask) {
  int t = blockIdx.x * blockDim.x + threadIdx.x;
  if (t >= PRE_NMS * MASK_WORDS) return;
  int i = t / MASK_WORDS, w = t % MASK_WORDS;
  int jbase = w * 32;
  if (jbase + 31 <= i) { mask[t] = 0u; return; }  // whole word below diagonal
  float x1i = sboxes[(size_t)i * 4 + 0];
  float y1i = sboxes[(size_t)i * 4 + 1];
  float x2i = sboxes[(size_t)i * 4 + 2];
  float y2i = sboxes[(size_t)i * 4 + 3];
  float ai = (x2i - x1i + 1.0f) * (y2i - y1i + 1.0f);
  uint32_t m = 0u;
  for (int b = 0; b < 32; ++b) {
    int j = jbase + b;
    if (j <= i) continue;                 // only suppress later boxes
    float bx1 = sboxes[(size_t)j * 4 + 0];
    float by1 = sboxes[(size_t)j * 4 + 1];
    float bx2 = sboxes[(size_t)j * 4 + 2];
    float by2 = sboxes[(size_t)j * 4 + 3];
    float xx1 = fmaxf(x1i, bx1), yy1 = fmaxf(y1i, by1);
    float xx2 = fminf(x2i, bx2), yy2 = fminf(y2i, by2);
    float ww = fmaxf(xx2 - xx1 + 1.0f, 0.0f);
    float hh = fmaxf(yy2 - yy1 + 1.0f, 0.0f);
    float inter = ww * hh;
    float aj = (bx2 - bx1 + 1.0f) * (by2 - by1 + 1.0f);
    float iou = inter / (ai + aj - inter);
    if (iou > NMS_THRESH) m |= (1u << b);
  }
  mask[t] = m;
}

// ---------------- phase B: greedy scan, removed-mask in registers (1 wave) ----------------
__global__ __launch_bounds__(32) void k_nms_scan(const uint32_t* __restrict__ mask,
                                                 const int* __restrict__ svalid,
                                                 const float* __restrict__ sboxes,
                                                 float* __restrict__ out) {
  const int lane = threadIdx.x;
  // 375 words distributed: lane L owns words w = g*32 + L (g = 0..11)
  uint32_t rem[12];
#pragma unroll
  for (int g = 0; g < 12; ++g) rem[g] = 0u;
  int cnt = 0;

#pragma unroll
  for (int g = 0; g < 12; ++g) {          // i in [g*1024, g*1024+1024): word idx == g
    int iend = (g + 1) * 1024;
    if (iend > PRE_NMS) iend = PRE_NMS;
    for (int i = g * 1024; i < iend; ++i) {
      uint32_t rw = (uint32_t)__shfl((int)rem[g], (i >> 5) & 31, 32);
      bool sup = (rw >> (i & 31)) & 1u;
      if (!sup && svalid[i]) {
        if (cnt < POST_NMS && lane == 0) {
          out[OUT_ROI + (size_t)cnt * 4 + 0] = sboxes[(size_t)i * 4 + 0];
          out[OUT_ROI + (size_t)cnt * 4 + 1] = sboxes[(size_t)i * 4 + 1];
          out[OUT_ROI + (size_t)cnt * 4 + 2] = sboxes[(size_t)i * 4 + 2];
          out[OUT_ROI + (size_t)cnt * 4 + 3] = sboxes[(size_t)i * 4 + 3];
        }
        ++cnt;
        const uint32_t* row = mask + (size_t)i * MASK_WORDS;
#pragma unroll
        for (int idx = 0; idx < 12; ++idx) {
          int w2 = idx * 32 + lane;
          if (w2 < MASK_WORDS) rem[idx] |= row[w2];
        }
      }
    }
  }
}

// ---------------- launcher ----------------
extern "C" void kernel_launch(void* const* d_in, const int* in_sizes, int n_in,
                              void* d_out, int out_size, void* d_ws, size_t ws_size,
                              hipStream_t stream) {
  const float* x       = (const float*)d_in[0];
  const int*   img_h   = (const int*)d_in[1];
  const int*   img_w   = (const int*)d_in[2];
  const int*   scale   = (const int*)d_in[3];
  const float* w1      = (const float*)d_in[4];
  const float* b1      = (const float*)d_in[5];
  const float* w_loc   = (const float*)d_in[6];
  const float* b_loc   = (const float*)d_in[7];
  const float* w_score = (const float*)d_in[8];
  const float* b_score = (const float*)d_in[9];
  float* out = (float*)d_out;

  char* ws = (char*)d_ws;
  // Region A [0, 10.24 MB): xh during GEMM phase; proposal buffers afterwards.
  _Float16* xh     = (_Float16*)(ws + 0);
  float*    boxes  = (float*)(ws + 0);             // 1,440,000 B (aliases xh)
  float*    key    = (float*)(ws + 1440000);       //   360,000 B
  float*    sboxes = (float*)(ws + 1800000);       //   192,000 B
  int*      svalid = (int*)(ws + 1992000);         //    48,000 B
  // Region B/C [10.24 MB, 25.26 MB): packed weights + h1h; dead before NMS.
  _Float16* whp    = (_Float16*)(ws + 10240000);   // 4,718,592 B
  _Float16* whd    = (_Float16*)(ws + 14958592);   //    65,536 B
  _Float16* h1h    = (_Float16*)(ws + 15024128);   // 10,240,000 B
  uint32_t* mask   = (uint32_t*)(ws + 10240000);   // 18,000,000 B (aliases B/C)
  // total workspace requirement: 28,240,000 B

  // 1) conversions / packing
  k_convert_x<<<(NPIX * C + 255) / 256, 256, 0, stream>>>(x, xh);
  k_pack_w1<<<(9 * 16 * 32 * 512 + 255) / 256, 256, 0, stream>>>(w1, whp);
  k_pack_whead<<<(16 * 4 * 512 + 255) / 256, 256, 0, stream>>>(w_loc, w_score, whd);

  // 2) anchors + roi zero-init
  k_anchors<<<(NBOX + 255) / 256, 256, 0, stream>>>(out);
  k_zero_roi<<<(POST_NMS * 4 + 255) / 256, 256, 0, stream>>>(out);

  // 3) 3x3 conv (WMMA implicit GEMM): 79 M-tiles x 8 N-tiles
  dim3 cgrid(79, 8);
  k_conv<<<cgrid, 256, 0, stream>>>(xh, whp, b1, h1h);

  // 4) head GEMM -> rpn_locs / rpn_scores in d_out
  k_head<<<79, 256, 0, stream>>>(h1h, whd, b_loc, b_score, out);

  // 5) decode + mask
  k_decode<<<(NBOX + 255) / 256, 256, 0, stream>>>(out, img_h, img_w, scale, boxes, key);

  // 6) exact stable descending top-PRE_NMS (LDS-tiled rank counting)
  k_rank<<<(NBOX + 255) / 256, 256, 0, stream>>>(key, boxes, sboxes, svalid);

  // 7) NMS phase A: suppression bit-matrix (parallel over 4.5M (row,word) pairs)
  k_mask<<<(PRE_NMS * MASK_WORDS + 255) / 256, 256, 0, stream>>>(sboxes, mask);

  // 8) NMS phase B: greedy scan with register-resident removed mask + gather
  k_nms_scan<<<1, 32, 0, stream>>>(mask, svalid, sboxes, out);
}
